// StdKAN_4887672783480
// MI455X (gfx1250) — matrix operand order repaired and checked by reading, not compiled, not run on hardware
//
#include <hip/hip_runtime.h>
#include <stdint.h>

// ---------------------------------------------------------------------------
// StdKAN (3 layers) for gfx1250.
//   B=4096, D=512, G=8, K=3  -> 11 basis fns, grid h=0.25, knots -1.75..1.75
// Batch-norm is folded into the weights so the whole layer collapses to one
// bf16 WMMA GEMM:  Out[b,o] = sum_k A[b,k]*Wt[o,k] + bias[o]
//   A   : (4096 x 6144) bf16, kappa = i*12 + k   (k=0..10 basis, k=11 silu)
//   Wt  : (512  x 6144) bf16, K-major rows       (c_basis/std  ||  c_res)
//   bias: -sum_ik mean*c_basis/std
// ---------------------------------------------------------------------------

#define BATCH 4096
#define DIM   512
#define COEF  11
#define PACKW 12
#define KTOT  (DIM * PACKW)   // 6144
#define STATN (DIM * COEF)    // 5632
#define KORD  3
#define HSTEP 0.25f
#define EPSV  1e-5f

typedef __attribute__((ext_vector_type(16))) __bf16 v16bf;
typedef __attribute__((ext_vector_type(8)))  float  v8f;

union Frag16 { uint4 u[2]; v16bf v; };

__device__ __forceinline__ unsigned short f2bf(float f) {
  uint32_t u = __float_as_uint(f);
  u += 0x7fffu + ((u >> 16) & 1u);            // round-to-nearest-even
  return (unsigned short)(u >> 16);
}

// ---------------------------------------------------------------------------
// Kernel 1: spline basis + silu -> packed bf16 A, plus per-(i,k) sum / sumsq.
// Block = 256 threads = 32 columns x 8 row-waves; each thread walks 32 rows.
// Coalesced X reads (consecutive tx -> consecutive columns of a row).
// ---------------------------------------------------------------------------
__global__ __launch_bounds__(256) void kan_basis_stats(
    const float* __restrict__ X, unsigned short* __restrict__ A,
    float* __restrict__ sum, float* __restrict__ sumsq)
{
  const int tx = threadIdx.x & 31;
  const int wv = threadIdx.x >> 5;            // 8 waves, one row-group each
  const int i  = blockIdx.x * 32 + tx;        // input feature
  const int r0 = blockIdx.y * 256 + wv * 32;  // first row for this wave

  float s1[COEF], s2[COEF];
  #pragma unroll
  for (int k = 0; k < COEF; ++k) { s1[k] = 0.0f; s2[k] = 0.0f; }

  for (int rr = 0; rr < 32; ++rr) {
    const int row = r0 + rr;
    const float x = X[(size_t)row * DIM + i];

    // Cox-de Boor, uniform knots t_j = -1 + (j-3)*h, j = 0..14
    float b[14];
    #pragma unroll
    for (int j = 0; j < 14; ++j) {
      const float t0 = -1.0f + (float)(j - KORD) * HSTEP;
      b[j] = (x >= t0 && x < t0 + HSTEP) ? 1.0f : 0.0f;
    }
    #pragma unroll
    for (int p = 1; p <= KORD; ++p) {
      const float inv = 1.0f / ((float)p * HSTEP);
      #pragma unroll
      for (int j = 0; j < 14 - p; ++j) {
        const float tj   = -1.0f + (float)(j - KORD) * HSTEP;
        const float tjp1 = tj + (float)(p + 1) * HSTEP;
        b[j] = (x - tj) * inv * b[j] + (tjp1 - x) * inv * b[j + 1];
      }
    }
    const float sil = x / (1.0f + __expf(-x));

    unsigned short pk[PACKW];
    #pragma unroll
    for (int k = 0; k < COEF; ++k) {
      pk[k] = f2bf(b[k]);
      s1[k] += b[k];
      s2[k] += b[k] * b[k];
    }
    pk[11] = f2bf(sil);

    // 24 contiguous bytes, 8-byte aligned (kappa base = i*12)
    uint2* dst = (uint2*)(A + (size_t)row * KTOT + (size_t)i * PACKW);
    dst[0] = make_uint2((uint32_t)pk[0] | ((uint32_t)pk[1] << 16),
                        (uint32_t)pk[2] | ((uint32_t)pk[3] << 16));
    dst[1] = make_uint2((uint32_t)pk[4] | ((uint32_t)pk[5] << 16),
                        (uint32_t)pk[6] | ((uint32_t)pk[7] << 16));
    dst[2] = make_uint2((uint32_t)pk[8] | ((uint32_t)pk[9] << 16),
                        (uint32_t)pk[10] | ((uint32_t)pk[11] << 16));
  }

  // cross-wave reduction of the 256-row partials, then one atomic per stat
  __shared__ float lsum[8][32][COEF];
  __shared__ float lsq [8][32][COEF];
  #pragma unroll
  for (int k = 0; k < COEF; ++k) { lsum[wv][tx][k] = s1[k]; lsq[wv][tx][k] = s2[k]; }
  __syncthreads();

  for (int idx = threadIdx.x; idx < 32 * COEF; idx += 256) {
    const int c = idx / COEF, k = idx - c * COEF;
    float a = 0.0f, q = 0.0f;
    #pragma unroll
    for (int w = 0; w < 8; ++w) { a += lsum[w][c][k]; q += lsq[w][c][k]; }
    const int gi = (blockIdx.x * 32 + c) * COEF + k;
    atomicAdd(&sum[gi], a);
    atomicAdd(&sumsq[gi], q);
  }
}

// ---------------------------------------------------------------------------
// Kernel 2: finalize batch statistics.
// ---------------------------------------------------------------------------
__global__ __launch_bounds__(256) void kan_finalize(
    const float* __restrict__ sum, const float* __restrict__ sumsq,
    float* __restrict__ meanv, float* __restrict__ invstd)
{
  const int idx = blockIdx.x * 256 + threadIdx.x;
  if (idx < STATN) {
    const float m = sum[idx] * (1.0f / (float)BATCH);
    const float v = sumsq[idx] * (1.0f / (float)BATCH) - m * m;
    meanv[idx]  = m;
    invstd[idx] = rsqrtf(fmaxf(v, 0.0f) + EPSV);
  }
}

// ---------------------------------------------------------------------------
// Kernel 3: fold norm into weights -> bf16 Wt (K-major rows) + fp32 bias.
// One block per output neuron o.
// ---------------------------------------------------------------------------
__global__ __launch_bounds__(256) void kan_prep_weights(
    const float* __restrict__ c_basis, const float* __restrict__ c_res,
    const float* __restrict__ meanv, const float* __restrict__ invstd,
    unsigned short* __restrict__ Wt, float* __restrict__ bias)
{
  const int o = blockIdx.x;
  float bacc = 0.0f;
  for (int kk = threadIdx.x; kk < KTOT; kk += 256) {
    const int i = kk / PACKW, k = kk - i * PACKW;
    float w;
    if (k < COEF) {
      const int s = i * COEF + k;
      w = c_basis[((size_t)o * DIM + i) * COEF + k] * invstd[s];
      bacc -= meanv[s] * w;
    } else {
      w = c_res[(size_t)o * DIM + i];
    }
    Wt[(size_t)o * KTOT + kk] = f2bf(w);
  }
  // block reduction of the folded bias term
  __shared__ float red[8];
  float v = bacc;
  #pragma unroll
  for (int d = 16; d > 0; d >>= 1) v += __shfl_down(v, d, 32);
  if ((threadIdx.x & 31) == 0) red[threadIdx.x >> 5] = v;
  __syncthreads();
  if (threadIdx.x == 0) {
    float t = 0.0f;
    #pragma unroll
    for (int w = 0; w < 8; ++w) t += red[w];
    bias[o] = t;
  }
}

// ---------------------------------------------------------------------------
// Kernel 4: Out = A(4096x6144,bf16) * Wt^T(512x6144,bf16) + bias, fp32 out.
// Block tile 128x128 (8 waves 4x2), wave tile 32x64 -> 2x4 v8f accumulators.
// Fragment loads follow the CDNA5 16-bit WMMA VGPR layouts:
//   A lane(l): row m0+l%16; K runs [k0+8h .. +7] and [k0+16+8h .. +7]
//   B lane(l): col n0+l%16; K runs [k0+8h .. +7] and [k0+16+8h .. +7]   (Wt is K-major)
// ---------------------------------------------------------------------------
#define BM 128
#define BN 128
#define WMT 32
#define WNT 64

__global__ __launch_bounds__(256) void kan_gemm_wmma(
    const unsigned short* __restrict__ A, const unsigned short* __restrict__ Wt,
    const float* __restrict__ bias, float* __restrict__ Out)
{
  const int lane = threadIdx.x & 31;
  const int wv   = threadIdx.x >> 5;
  const int wm   = wv & 3;               // 0..3 (M)
  const int wn   = wv >> 2;              // 0..1 (N)
  const int hf   = lane >> 4;            // lane half
  const int l16  = lane & 15;

  const int m0 = blockIdx.x * BM + wm * WMT;
  const int n0 = blockIdx.y * BN + wn * WNT;

  v8f acc[2][4];
  const v8f vz = {0.f, 0.f, 0.f, 0.f, 0.f, 0.f, 0.f, 0.f};
  #pragma unroll
  for (int a = 0; a < 2; ++a)
    #pragma unroll
    for (int b = 0; b < 4; ++b) acc[a][b] = vz;

  const unsigned short* arow[2];
  arow[0] = A + (size_t)(m0 + l16) * KTOT;
  arow[1] = arow[0] + (size_t)16 * KTOT;
  const unsigned short* brow[4];
  #pragma unroll
  for (int nf = 0; nf < 4; ++nf)
    brow[nf] = Wt + (size_t)(n0 + nf * 16 + l16) * KTOT;

  for (int k0 = 0; k0 < KTOT; k0 += 32) {
    Frag16 af[2], bfr[4];
    #pragma unroll
    for (int mf = 0; mf < 2; ++mf) {
      af[mf].u[0] = *(const uint4*)(arow[mf] + k0 + 8 * hf);
      af[mf].u[1] = *(const uint4*)(arow[mf] + k0 + 16 + 8 * hf);
    }
    #pragma unroll
    for (int nf = 0; nf < 4; ++nf) {
      bfr[nf].u[0] = *(const uint4*)(brow[nf] + k0 + 8 * hf);
      bfr[nf].u[1] = *(const uint4*)(brow[nf] + k0 + 16 + 8 * hf);
    }
    // stream-ahead on the activation rows (emits global_prefetch_b8)
    __builtin_prefetch((const char*)(arow[0] + k0 + 256), 0, 1);
    __builtin_prefetch((const char*)(arow[1] + k0 + 256), 0, 1);

    #pragma unroll
    for (int mf = 0; mf < 2; ++mf)
      #pragma unroll
      for (int nf = 0; nf < 4; ++nf)
        acc[mf][nf] = __builtin_amdgcn_wmma_f32_16x16x32_bf16(
            false, af[mf].v, false, bfr[nf].v,
            (short)0, acc[mf][nf], false, false);
  }

  #pragma unroll
  for (int nf = 0; nf < 4; ++nf) {
    const int n = n0 + nf * 16 + l16;
    const float bv = bias[n];
    #pragma unroll
    for (int mf = 0; mf < 2; ++mf) {
      #pragma unroll
      for (int r = 0; r < 8; ++r) {
        const int m = m0 + mf * 16 + hf * 8 + r;
        Out[(size_t)m * DIM + n] = acc[mf][nf][r] + bv;
      }
    }
  }
}

// ---------------------------------------------------------------------------
// Host orchestration: 3 layers, all on `stream` (graph-capture safe).
// Workspace: A(48MB bf16) | Wt(6MB bf16) | stats | mean/invstd | bias | H(8MB)
// ---------------------------------------------------------------------------
extern "C" void kernel_launch(void* const* d_in, const int* in_sizes, int n_in,
                              void* d_out, int out_size, void* d_ws, size_t ws_size,
                              hipStream_t stream) {
  (void)in_sizes; (void)n_in; (void)out_size; (void)ws_size;

  const float* X = (const float*)d_in[0];
  const float* cres[3]   = {(const float*)d_in[1], (const float*)d_in[3], (const float*)d_in[5]};
  const float* cbasis[3] = {(const float*)d_in[2], (const float*)d_in[4], (const float*)d_in[6]};

  char* p = (char*)d_ws;
  unsigned short* A  = (unsigned short*)p; p += (size_t)BATCH * KTOT * 2;
  unsigned short* Wt = (unsigned short*)p; p += (size_t)DIM * KTOT * 2;
  float* sum    = (float*)p; p += (size_t)STATN * 4;
  float* sumsq  = (float*)p; p += (size_t)STATN * 4;
  float* meanv  = (float*)p; p += (size_t)STATN * 4;
  float* invstd = (float*)p; p += (size_t)STATN * 4;
  float* bias   = (float*)p; p += (size_t)DIM * 4;
  float* H      = (float*)p; p += (size_t)BATCH * DIM * 4;

  for (int L = 0; L < 3; ++L) {
    const float* in = (L == 0) ? X : H;
    float* out = (L == 2) ? (float*)d_out : H;

    hipMemsetAsync(sum, 0, (size_t)2 * STATN * sizeof(float), stream); // sum+sumsq

    kan_basis_stats<<<dim3(DIM / 32, BATCH / 256), 256, 0, stream>>>(in, A, sum, sumsq);
    kan_finalize<<<(STATN + 255) / 256, 256, 0, stream>>>(sum, sumsq, meanv, invstd);
    kan_prep_weights<<<DIM, 256, 0, stream>>>(cbasis[L], cres[L], meanv, invstd, Wt, bias);
    kan_gemm_wmma<<<dim3(BATCH / BM, DIM / BN), 256, 0, stream>>>(A, Wt, bias, out);
  }
}
